// Block_38620345926379
// MI455X (gfx1250) — compile-verified, matrix operand order checked by reading, not run on hardware
//
#include <hip/hip_runtime.h>
#include <hip/hip_bf16.h>

// ---------------------------------------------------------------------------
// Transformer block for MI455X (gfx1250, wave32, WMMA + TDM).
// Shapes: B=256, S=256, E=384, H=6, DH=64, 4E=1536, tokens M=65536.
// GEMMs: TDM (tensor_load_to_lds) double-buffered LDS staging + WMMA bf16.
// ---------------------------------------------------------------------------

#define TOKENS   65536      // B*S
#define EMB      384
#define EMB4     1536
#define NHEAD    6
#define DHEAD    64
#define SEQ      256
#define QKVN     1152       // 3*EMB

// GEMM tiling
#define KC        128                    // K chunk staged per TDM load
#define LDS_ROW   136                    // elements: 128 + 8 pad (272B, bank-spread)
#define A_CHUNK_E (64 * LDS_ROW)         // 8704 elements  (17408 B)
#define B_CHUNK_E (128 * LDS_ROW)        // 17408 elements (34816 B)
#define LDS_A0_E  0
#define LDS_A1_E  (A_CHUNK_E)
#define LDS_B0_E  (2 * A_CHUNK_E)
#define LDS_B1_E  (2 * A_CHUNK_E + B_CHUNK_E)
#define GEMM_LDS_BYTES ((2 * A_CHUNK_E + 2 * B_CHUNK_E) * 2)   // 104448

typedef __attribute__((ext_vector_type(16))) __bf16 bf16x16;
typedef __attribute__((ext_vector_type(8)))  float  f32x8;
typedef __attribute__((ext_vector_type(4)))  unsigned int u32x4;
typedef __attribute__((ext_vector_type(8)))  int i32x8;
typedef __attribute__((ext_vector_type(4)))  int i32x4;

union FragU {
    uint4   u[2];
    bf16x16 v;
};

__device__ __forceinline__ f32x8 zero8() {
    f32x8 z = {0.f, 0.f, 0.f, 0.f, 0.f, 0.f, 0.f, 0.f};
    return z;
}

__device__ __forceinline__ unsigned short f2bf(float f) {
    unsigned u = __float_as_uint(f);
    unsigned r = (u + 0x7FFFu + ((u >> 16) & 1u)) >> 16;
    return (unsigned short)r;
}

// Load a 16-bit WMMA fragment whose K-values are contiguous at `p`.
// Element e <-> K = klo + e (e<8) or klo + e + 8 (e>=8); klo = (lane>=16)?8:0.
__device__ __forceinline__ bf16x16 load_frag(const unsigned short* p, int klo) {
    FragU f;
    f.u[0] = *reinterpret_cast<const uint4*>(p + klo);
    f.u[1] = *reinterpret_cast<const uint4*>(p + klo + 16);
    return f.v;
}

__device__ __forceinline__ f32x8 wmma_bf16(bf16x16 a, bf16x16 b, f32x8 c) {
    return __builtin_amdgcn_wmma_f32_16x16x32_bf16(
        false, a, false, b, (short)0, c, false, false);
}

// ---------------------------------------------------------------------------
// TDM: async 2-D tile load (rows x KC bf16) global -> LDS with row padding.
// tensor == tile (always in-bounds), dim0 stride = k_stride elements.
// LDS pad: every 64 DWORDs (one 256B row) insert 4 DWORDs -> 272B row stride.
// ---------------------------------------------------------------------------
__device__ __forceinline__ void tdm_load_2d(const unsigned short* gptr,
                                            unsigned lds_byte_off,
                                            unsigned rows, unsigned k_stride) {
    unsigned long long ga = (unsigned long long)(size_t)gptr;
    u32x4 g0;
    g0[0] = 1u;                                     // count=1, user mode
    g0[1] = lds_byte_off;                           // lds_addr
    g0[2] = (unsigned)(ga & 0xFFFFFFFFu);           // global_addr[31:0]
    g0[3] = (unsigned)((ga >> 32) & 0x01FFFFFFu)    // global_addr[56:32]
          | 0x80000000u;                            // type=2 (image)
    i32x8 g1;
    g1[0] = (int)((1u << 16)                        // data_size = 2 bytes
                | (1u << 20)                        // pad_enable
                | (5u << 22)                        // pad_interval: 64 DWORDs
                | (3u << 25));                      // pad_amount: 4 DWORDs
    g1[1] = (int)((KC & 0xFFFFu) << 16);            // tensor_dim0[15:0] @ bits 63:48
    g1[2] = (int)((KC >> 16) | ((rows & 0xFFFFu) << 16)); // t_dim0 hi | t_dim1 lo
    g1[3] = (int)((rows >> 16) | (KC << 16));       // t_dim1 hi | tile_dim0 = KC
    g1[4] = (int)rows;                              // tile_dim1 = rows, tile_dim2 = 0
    g1[5] = (int)k_stride;                          // tensor_dim0_stride[31:0]
    g1[6] = 0;                                      // stride hi | dim1_stride lo
    g1[7] = 0;
    i32x4 gz = {0, 0, 0, 0};
#if __has_include(<hip/amd_detail/amd_gfx1250_TDM.h>)
    i32x8 gz8 = {0, 0, 0, 0, 0, 0, 0, 0};
    __builtin_amdgcn_tensor_load_to_lds(g0, g1, gz, gz, gz8, 0);
#else
    __builtin_amdgcn_tensor_load_to_lds(g0, g1, gz, gz, 0);
#endif
}

// ---------------------------------------------------------------------------
// Weight preparation: bf16 + transpose to [N][K] (K-contiguous) layout
// ---------------------------------------------------------------------------
__global__ void conv_qkv_w_kernel(const float* __restrict__ Wq,
                                  const float* __restrict__ Wk,
                                  const float* __restrict__ Wv,
                                  unsigned short* __restrict__ out) {
    int idx = blockIdx.x * blockDim.x + threadIdx.x;
    if (idx >= QKVN * EMB) return;
    int n = idx / EMB;
    int e = idx % EMB;
    int g = n / EMB;          // 0=q 1=k 2=v
    int wi = n % EMB;
    int h = wi / DHEAD;
    int d = wi % DHEAD;
    const float* src = (g == 0) ? Wq : ((g == 1) ? Wk : Wv);
    float v = src[((size_t)h * EMB + e) * DHEAD + d];
    if (g == 0) v *= 0.125f;  // 1/sqrt(DH)
    out[idx] = f2bf(v);
}

__global__ void conv_transpose_kernel(const float* __restrict__ W,
                                      unsigned short* __restrict__ Wt,
                                      int K, int N) {
    int idx = blockIdx.x * blockDim.x + threadIdx.x;
    if (idx >= K * N) return;
    int n = idx / K;
    int k = idx % K;
    Wt[idx] = f2bf(W[(size_t)k * N + n]);
}

// ---------------------------------------------------------------------------
// LayerNorm (f32 in -> bf16 out), one wave per row of 384
// ---------------------------------------------------------------------------
__global__ __launch_bounds__(256) void ln_kernel(const float* __restrict__ x,
                                                 const float* __restrict__ g,
                                                 const float* __restrict__ bta,
                                                 unsigned short* __restrict__ out) {
    const int w = threadIdx.x >> 5;
    const int lane = threadIdx.x & 31;
    const size_t row = (size_t)blockIdx.x * 8 + w;
    const float* xr = x + row * EMB;
    float v[12];
    float s = 0.f;
#pragma unroll
    for (int j = 0; j < 12; ++j) { v[j] = xr[lane + j * 32]; s += v[j]; }
#pragma unroll
    for (int m = 16; m; m >>= 1) s += __shfl_xor(s, m);
    float mu = s * (1.0f / EMB);
    float q = 0.f;
#pragma unroll
    for (int j = 0; j < 12; ++j) { float d = v[j] - mu; q += d * d; }
#pragma unroll
    for (int m = 16; m; m >>= 1) q += __shfl_xor(q, m);
    float rs = rsqrtf(q * (1.0f / EMB) + 1e-5f);
    unsigned short* orow = out + row * EMB;
#pragma unroll
    for (int j = 0; j < 12; ++j) {
        int c = lane + j * 32;
        orow[c] = f2bf((v[j] - mu) * rs * g[c] + bta[c]);
    }
}

// ---------------------------------------------------------------------------
// TDM-staged bf16 WMMA GEMM: C[M,N] = A[M,K]*B[K,N] (+bias)(+resid)(relu)
// A row-major bf16; Bt = B^T [N,K] bf16 (K-contiguous).
// Block: 256 thr = 8 waves (2x4); wave 32x32; block 64x128; K staged in
// double-buffered LDS chunks of 128 via tensor_load_to_lds (wave 0 issues).
// ---------------------------------------------------------------------------
__device__ __forceinline__ void store_tile(f32x8 c, int row0, int col0, int lane,
                                           const float* bias, const float* resid,
                                           float* outF, unsigned short* outB,
                                           int N, int relu) {
    const int lcol = lane & 15;
    const int lh = lane >> 4;
    const int col = col0 + lcol;
    const float bv = bias ? bias[col] : 0.f;
#pragma unroll
    for (int r = 0; r < 8; ++r) {
        int row = row0 + r + lh * 8;
        float v = c[r] + bv;
        if (resid) v += resid[(size_t)row * N + col];
        if (relu) v = fmaxf(v, 0.f);
        if (outF) outF[(size_t)row * N + col] = v;
        else      outB[(size_t)row * N + col] = f2bf(v);
    }
}

__global__ __launch_bounds__(256) void gemm_kernel(const unsigned short* __restrict__ A,
                                                   const unsigned short* __restrict__ Bt,
                                                   const float* __restrict__ bias,
                                                   const float* __restrict__ resid,
                                                   float* __restrict__ outF,
                                                   unsigned short* __restrict__ outB,
                                                   int N, int K, int relu) {
    extern __shared__ unsigned short smem[];
    const int lane = threadIdx.x & 31;
    const int w = threadIdx.x >> 5;
    const int lcol = lane & 15;
    const int klo = (lane >> 4) * 8;
    const int rowBase = blockIdx.x * 64;            // block M origin
    const int colBase = blockIdx.y * 128;           // block N origin
    const int arow0 = (w >> 2) * 32 + lcol;         // wave A row in block tile
    const int brow0 = (w & 3) * 32 + lcol;          // wave B row (=col) in tile

    const int nch = K / KC;

    // prologue: issue chunk 0 (and 1) - wave 0 only
    if (w == 0) {
        tdm_load_2d(A + (size_t)rowBase * K, LDS_A0_E * 2, 64, (unsigned)K);
        tdm_load_2d(Bt + (size_t)colBase * K, LDS_B0_E * 2, 128, (unsigned)K);
        if (nch > 1) {
            tdm_load_2d(A + (size_t)rowBase * K + KC, LDS_A1_E * 2, 64, (unsigned)K);
            tdm_load_2d(Bt + (size_t)colBase * K + KC, LDS_B1_E * 2, 128, (unsigned)K);
        }
    }

    f32x8 c00 = zero8(), c01 = zero8(), c10 = zero8(), c11 = zero8();

    for (int c = 0; c < nch; ++c) {
        const int cur = c & 1;
        if (w == 0) {
            if (c + 1 < nch)
                __builtin_amdgcn_s_wait_tensorcnt((short)2);  // next chunk still in flight
            else
                __builtin_amdgcn_s_wait_tensorcnt((short)0);  // drain everything
        }
        __syncthreads();

        const unsigned short* As = smem + (cur ? LDS_A1_E : LDS_A0_E);
        const unsigned short* Bs = smem + (cur ? LDS_B1_E : LDS_B0_E);
        const unsigned short* a0p = As + arow0 * LDS_ROW;
        const unsigned short* a1p = As + (arow0 + 16) * LDS_ROW;
        const unsigned short* b0p = Bs + brow0 * LDS_ROW;
        const unsigned short* b1p = Bs + (brow0 + 16) * LDS_ROW;
#pragma unroll
        for (int kk = 0; kk < KC; kk += 32) {
            bf16x16 a0 = load_frag(a0p + kk, klo);
            bf16x16 a1 = load_frag(a1p + kk, klo);
            bf16x16 b0 = load_frag(b0p + kk, klo);
            bf16x16 b1 = load_frag(b1p + kk, klo);
            c00 = wmma_bf16(a0, b0, c00);
            c01 = wmma_bf16(a0, b1, c01);
            c10 = wmma_bf16(a1, b0, c10);
            c11 = wmma_bf16(a1, b1, c11);
        }
        __syncthreads();

        // refill the buffer just consumed with chunk c+2
        if (w == 0 && c + 2 < nch) {
            const int kc2 = (c + 2) * KC;
            tdm_load_2d(A + (size_t)rowBase * K + kc2,
                        (cur ? LDS_A1_E : LDS_A0_E) * 2, 64, (unsigned)K);
            tdm_load_2d(Bt + (size_t)colBase * K + kc2,
                        (cur ? LDS_B1_E : LDS_B0_E) * 2, 128, (unsigned)K);
        }
    }

    const int wr = rowBase + (w >> 2) * 32;
    const int wc = colBase + (w & 3) * 32;
    store_tile(c00, wr,      wc,      lane, bias, resid, outF, outB, N, relu);
    store_tile(c01, wr,      wc + 16, lane, bias, resid, outF, outB, N, relu);
    store_tile(c10, wr + 16, wc,      lane, bias, resid, outF, outB, N, relu);
    store_tile(c11, wr + 16, wc + 16, lane, bias, resid, outF, outB, N, relu);
}

// ---------------------------------------------------------------------------
// V transpose: qkv[:, 768 + h*64 + d] -> vT[(b*H+h)*64 + d][t]  (bf16)
// ---------------------------------------------------------------------------
__global__ __launch_bounds__(256) void transpose_v_kernel(const unsigned short* __restrict__ qkv,
                                                          unsigned short* __restrict__ vT) {
    const int bh = blockIdx.x;
    const int b = bh / NHEAD;
    const int h = bh % NHEAD;
    __shared__ unsigned short tile[64][72];
    for (int tc = 0; tc < 4; ++tc) {
        const int t0 = tc * 64;
        {
            int t = threadIdx.x >> 2;
            int dg = threadIdx.x & 3;
            const unsigned short* src =
                qkv + (size_t)(b * SEQ + t0 + t) * QKVN + 2 * EMB + h * DHEAD + dg * 16;
            *reinterpret_cast<uint4*>(&tile[t][dg * 16]) =
                *reinterpret_cast<const uint4*>(src);
            *reinterpret_cast<uint4*>(&tile[t][dg * 16 + 8]) =
                *reinterpret_cast<const uint4*>(src + 8);
        }
        __syncthreads();
        {
            int d = threadIdx.x >> 2;
            int tg = threadIdx.x & 3;
            alignas(16) unsigned short tmp[16];
#pragma unroll
            for (int j = 0; j < 16; ++j) tmp[j] = tile[tg * 16 + j][d];
            unsigned short* dst =
                vT + ((size_t)(b * NHEAD + h) * DHEAD + d) * SEQ + t0 + tg * 16;
            *reinterpret_cast<uint4*>(dst) = *reinterpret_cast<uint4*>(&tmp[0]);
            *reinterpret_cast<uint4*>(dst + 8) = *reinterpret_cast<uint4*>(&tmp[8]);
        }
        __syncthreads();
    }
}

// ---------------------------------------------------------------------------
// Causal attention: one wave per (b, h, 16-row q tile). 2 waves / block.
// ---------------------------------------------------------------------------
__global__ __launch_bounds__(64) void attn_kernel(const unsigned short* __restrict__ qkv,
                                                  const unsigned short* __restrict__ vT,
                                                  unsigned short* __restrict__ obuf) {
    const int lane = threadIdx.x & 31;
    const int w = threadIdx.x >> 5;
    const int lcol = lane & 15;
    const int lh = lane >> 4;
    const int klo = lh * 8;

    const int tile = blockIdx.x * 2 + w;
    const int b = tile / (NHEAD * 16);
    const int rem = tile % (NHEAD * 16);
    const int h = rem >> 4;
    const int i = rem & 15;
    const int nk = (i + 1) * 16;

    __shared__ float sc[2][16][SEQ];
    __shared__ float mrow[2][16];
    __shared__ float srow[2][16];

    const unsigned short* qrow =
        qkv + (size_t)(b * SEQ + i * 16 + lcol) * QKVN + h * DHEAD;
    bf16x16 qa0 = load_frag(qrow, klo);
    bf16x16 qa1 = load_frag(qrow + 32, klo);

    float pmax[8];
#pragma unroll
    for (int r = 0; r < 8; ++r) pmax[r] = -3.0e38f;

    for (int j = 0; j <= i; ++j) {
        const unsigned short* krow =
            qkv + (size_t)(b * SEQ + j * 16 + lcol) * QKVN + EMB + h * DHEAD;
        bf16x16 kb0 = load_frag(krow, klo);
        bf16x16 kb1 = load_frag(krow + 32, klo);
        f32x8 s = zero8();
        s = wmma_bf16(qa0, kb0, s);
        s = wmma_bf16(qa1, kb1, s);
        const int absc = j * 16 + lcol;
#pragma unroll
        for (int r = 0; r < 8; ++r) {
            int rowi = r + lh * 8;
            float v = (absc <= i * 16 + rowi) ? s[r] : -1.0e9f;
            pmax[r] = fmaxf(pmax[r], v);
            sc[w][rowi][absc] = v;
        }
    }
#pragma unroll
    for (int r = 0; r < 8; ++r) {
        float m = pmax[r];
        m = fmaxf(m, __shfl_xor(m, 1));
        m = fmaxf(m, __shfl_xor(m, 2));
        m = fmaxf(m, __shfl_xor(m, 4));
        m = fmaxf(m, __shfl_xor(m, 8));
        mrow[w][r + lh * 8] = m;
    }
    __syncthreads();

    {
        const int rowi = lane & 15;
        const float mr = mrow[w][rowi];
        float ssum = 0.f;
        for (int c = lane >> 4; c < nk; c += 2)
            ssum += __expf(sc[w][rowi][c] - mr);
        ssum += __shfl_xor(ssum, 16);
        srow[w][rowi] = 1.0f / ssum;
    }
    __syncthreads();

    const float mr2 = mrow[w][lcol];
    const float ir = srow[w][lcol];
    f32x8 o0 = zero8(), o1 = zero8(), o2 = zero8(), o3 = zero8();
    const size_t vbase = (size_t)(b * NHEAD + h) * DHEAD * SEQ;

    for (int jj = 0; jj < nk; jj += 32) {
        FragU pf;
        alignas(16) unsigned short pe[16];
#pragma unroll
        for (int e = 0; e < 16; ++e) {
            int k = (e < 8) ? (klo + e) : (klo + e + 8);
            int key = jj + k;
            float v = (key < nk) ? (__expf(sc[w][lcol][key] - mr2) * ir) : 0.f;
            pe[e] = f2bf(v);
        }
        pf.u[0] = *reinterpret_cast<uint4*>(&pe[0]);
        pf.u[1] = *reinterpret_cast<uint4*>(&pe[8]);
        bf16x16 pa = pf.v;

        const unsigned short* v0 = vT + vbase + (size_t)(lcol) * SEQ + jj;
        const unsigned short* v1 = vT + vbase + (size_t)(16 + lcol) * SEQ + jj;
        const unsigned short* v2 = vT + vbase + (size_t)(32 + lcol) * SEQ + jj;
        const unsigned short* v3 = vT + vbase + (size_t)(48 + lcol) * SEQ + jj;
        o0 = wmma_bf16(pa, load_frag(v0, klo), o0);
        o1 = wmma_bf16(pa, load_frag(v1, klo), o1);
        o2 = wmma_bf16(pa, load_frag(v2, klo), o2);
        o3 = wmma_bf16(pa, load_frag(v3, klo), o3);
    }

#pragma unroll
    for (int nc = 0; nc < 4; ++nc) {
        f32x8 oc = (nc == 0) ? o0 : (nc == 1) ? o1 : (nc == 2) ? o2 : o3;
#pragma unroll
        for (int r = 0; r < 8; ++r) {
            int rowi = r + lh * 8;
            size_t tok = (size_t)b * SEQ + i * 16 + rowi;
            obuf[tok * EMB + h * DHEAD + nc * 16 + lcol] = f2bf(oc[r]);
        }
    }
}

// ---------------------------------------------------------------------------
// Launch
// ---------------------------------------------------------------------------
extern "C" void kernel_launch(void* const* d_in, const int* in_sizes, int n_in,
                              void* d_out, int out_size, void* d_ws, size_t ws_size,
                              hipStream_t stream) {
    (void)in_sizes; (void)n_in; (void)out_size; (void)ws_size;
    const float* x   = (const float*)d_in[0];
    const float* Wq  = (const float*)d_in[1];
    const float* Wk  = (const float*)d_in[2];
    const float* Wv  = (const float*)d_in[3];
    const float* Wp  = (const float*)d_in[4];
    const float* bp  = (const float*)d_in[5];
    const float* W1  = (const float*)d_in[6];
    const float* b1  = (const float*)d_in[7];
    const float* W2  = (const float*)d_in[8];
    const float* b2  = (const float*)d_in[9];
    const float* g1  = (const float*)d_in[10];
    const float* be1 = (const float*)d_in[11];
    const float* g2  = (const float*)d_in[12];
    const float* be2 = (const float*)d_in[13];
    float* out = (float*)d_out;

    size_t cur = 0;
    auto alloc = [&](size_t bytes) -> void* {
        void* p = (char*)d_ws + cur;
        cur += (bytes + 255) & ~(size_t)255;
        return p;
    };
    unsigned short* wqkvT = (unsigned short*)alloc((size_t)QKVN * EMB * 2);
    unsigned short* wpT   = (unsigned short*)alloc((size_t)EMB * EMB * 2);
    unsigned short* w1T   = (unsigned short*)alloc((size_t)EMB4 * EMB * 2);
    unsigned short* w2T   = (unsigned short*)alloc((size_t)EMB * EMB4 * 2);
    unsigned short* hbf   = (unsigned short*)alloc((size_t)TOKENS * EMB * 2);   // h1 / h2
    unsigned short* qkv   = (unsigned short*)alloc((size_t)TOKENS * EMB4 * 2);  // qkv, later FFN1 act
    unsigned short* vT    = (unsigned short*)alloc((size_t)TOKENS * EMB * 2);
    unsigned short* obuf  = (unsigned short*)alloc((size_t)TOKENS * EMB * 2);
    float*          x2    = (float*)alloc((size_t)TOKENS * EMB * 4);
    unsigned short* act   = qkv;  // alias: qkv dead before FFN1 writes

    conv_qkv_w_kernel<<<(QKVN * EMB + 255) / 256, 256, 0, stream>>>(Wq, Wk, Wv, wqkvT);
    conv_transpose_kernel<<<(EMB * EMB + 255) / 256, 256, 0, stream>>>(Wp, wpT, EMB, EMB);
    conv_transpose_kernel<<<(EMB * EMB4 + 255) / 256, 256, 0, stream>>>(W1, w1T, EMB, EMB4);
    conv_transpose_kernel<<<(EMB4 * EMB + 255) / 256, 256, 0, stream>>>(W2, w2T, EMB4, EMB);

    // LN1
    ln_kernel<<<TOKENS / 8, 256, 0, stream>>>(x, g1, be1, hbf);
    // QKV GEMM: [65536,384] x [384,1152] -> bf16
    gemm_kernel<<<dim3(TOKENS / 64, QKVN / 128), 256, GEMM_LDS_BYTES, stream>>>(
        hbf, wqkvT, nullptr, nullptr, nullptr, qkv, QKVN, EMB, 0);
    // V transpose for PV fragment loads
    transpose_v_kernel<<<256 * NHEAD, 256, 0, stream>>>(qkv, vT);
    // causal attention
    attn_kernel<<<(256 * NHEAD * 16) / 2, 64, 0, stream>>>(qkv, vT, obuf);
    // output projection + bias + residual(x) -> x2 (f32)
    gemm_kernel<<<dim3(TOKENS / 64, EMB / 128), 256, GEMM_LDS_BYTES, stream>>>(
        obuf, wpT, bp, x, x2, nullptr, EMB, EMB, 0);
    // LN2
    ln_kernel<<<TOKENS / 8, 256, 0, stream>>>(x2, g2, be2, hbf);
    // FFN1 + bias + relu -> bf16
    gemm_kernel<<<dim3(TOKENS / 64, EMB4 / 128), 256, GEMM_LDS_BYTES, stream>>>(
        hbf, w1T, b1, nullptr, nullptr, act, EMB4, EMB, 1);
    // FFN2 + bias + residual(x2) -> out (f32)
    gemm_kernel<<<dim3(TOKENS / 64, EMB / 128), 256, GEMM_LDS_BYTES, stream>>>(
        act, w2T, b2, x2, out, nullptr, EMB, EMB4, 0);
}
